// TSModel_63410897158712
// MI455X (gfx1250) — compile-verified, matrix operand order checked by reading, not run on hardware
//
#include <hip/hip_runtime.h>
#include <math.h>

// ---- problem constants (match reference: B=512, HID=64) ----
#define HID   64
#define BATCH 512
#define ROWS  16            // batch rows per workgroup (one WMMA M-tile)
#define HP    72            // padded h1 row length in halves (bank-conflict free, 16B aligned)
#define GP    260           // padded gates row length in floats
#define CP    68            // padded c1 row length in floats (272B: 16B aligned, conflict-free)

typedef _Float16 half8  __attribute__((ext_vector_type(8)));
typedef _Float16 half16 __attribute__((ext_vector_type(16)));
typedef float    float8 __attribute__((ext_vector_type(8)));
typedef float    float4v __attribute__((ext_vector_type(4)));

// fast path: v_exp_f32 + v_rcp_f32 (args are >= 1, no div hazards possible)
__device__ __forceinline__ float sigf(float x) {
    return __builtin_amdgcn_rcpf(1.0f + __expf(-x));
}
__device__ __forceinline__ float tanhfast(float x) {
    return fmaf(2.0f, __builtin_amdgcn_rcpf(1.0f + __expf(-2.0f * x)), -1.0f);
}

__global__ __launch_bounds__(512, 1)
void lstm2_persistent(const float* __restrict__ x,      // [B,T]
                      const float* __restrict__ Wih1,   // [256,1]
                      const float* __restrict__ Whh1,   // [256,64]
                      const float* __restrict__ bih1,   // [256]
                      const float* __restrict__ bhh1,   // [256]
                      const float* __restrict__ Wih2,   // [4,64]
                      const float* __restrict__ Whh2,   // [4,1]
                      const float* __restrict__ bih2,   // [4]
                      const float* __restrict__ bhh2,   // [4]
                      float* __restrict__ out,          // [B,T] (emits c2)
                      int T)
{
    __shared__ _Float16 h1_lds[ROWS * HP];     // recurrent h1, f16 (WMMA A source)
    __shared__ float    gates_lds[ROWS * GP];  // layer-1 pre-activations [16 x 256]
    __shared__ float    c1_lds[ROWS * CP];     // layer-1 cell state (feeds layer 2)
    __shared__ float    wih2_lds[4 * HID];
    __shared__ float    xt_lds[ROWS];
    __shared__ float    h2_lds[ROWS];
    __shared__ float    g2_lds[ROWS * 4];

    const int tid   = threadIdx.x;
    const int lane  = tid & 31;
    const int w     = tid >> 5;              // wave id == N-tile id (16 gate columns)
    const int nl    = lane & 15;
    const int gcol  = w * 16 + nl;           // this lane's gate column (B/C/D N index)
    const int koff  = (lane < 16) ? 0 : 8;   // f16 fragment K sub-offset per lane half
    const int rbase = (lane < 16) ? 0 : 8;   // C/D M sub-offset per lane half
    const int brow  = blockIdx.x * ROWS;     // first batch row of this workgroup

    // ---- one-time: pack Whh1 into two f16 B-fragments (K=0..31, K=32..63), kept in VGPRs ----
    union { half16 v; _Float16 e[16]; } bf0, bf1;
    #pragma unroll
    for (int e = 0; e < 16; ++e) {
        int k0 = koff + e + ((e >= 8) ? 8 : 0);
        bf0.e[e] = (_Float16)Whh1[gcol * HID + k0];
        bf1.e[e] = (_Float16)Whh1[gcol * HID + 32 + k0];
    }
    const float wih1_s = Wih1[gcol];
    const float b1_s   = bih1[gcol] + bhh1[gcol];

    // layer-2 dot: thread = (m = tid>>5, q = (tid>>3)&3, kgrp = tid&7) -> 8 MACs each
    const int dm = tid >> 5;
    const int dq = (tid >> 3) & 3;
    const int dk = (tid & 7) * 8;
    const float whh2_s = Whh2[dq];
    const float b2_s   = bih2[dq] + bhh2[dq];

    // ---- zero-init recurrent state + preload constants and x(t=0) ----
    for (int i = tid; i < ROWS * HP; i += 512) h1_lds[i] = (_Float16)0.f;
    if (tid < 4 * HID) wih2_lds[tid] = Wih2[tid];
    if (tid < ROWS) { h2_lds[tid] = 0.f; xt_lds[tid] = x[(brow + tid) * T]; }
    float c1r0 = 0.f, c1r1 = 0.f;   // each thread owns two (m,h) layer-1 cells
    float c2r  = 0.f;               // threads < 16 own one layer-2 cell
    __syncthreads();

    const int em0 = tid >> 6,          eh0 = tid & 63;
    const int em1 = (tid + 512) >> 6,  eh1 = tid & 63;

    for (int t = 0; t < T; ++t) {
        // ---- layer 1: gates = x_t*Wih1 + b1 + h1 @ Whh1.T  (2 WMMA per wave) ----
        float8 c;
        #pragma unroll
        for (int r = 0; r < 8; ++r)
            c[r] = fmaf(xt_lds[r + rbase], wih1_s, b1_s);   // C init: bias + input term

        union { half16 v; half8 h[2]; } a0, a1;
        {
            const int base = (lane & 15) * HP + koff;       // M = lane%16 for both halves
            a0.h[0] = *(const half8*)&h1_lds[base];         // K =  0.. 7 (+koff)
            a0.h[1] = *(const half8*)&h1_lds[base + 16];    // K = 16..23 (+koff)
            a1.h[0] = *(const half8*)&h1_lds[base + 32];    // K = 32..39 (+koff)
            a1.h[1] = *(const half8*)&h1_lds[base + 48];    // K = 48..55 (+koff)
        }
        c = __builtin_amdgcn_wmma_f32_16x16x32_f16(false, a0.v, false, bf0.v, (short)0, c, false, false);
        c = __builtin_amdgcn_wmma_f32_16x16x32_f16(false, a1.v, false, bf1.v, (short)0, c, false, false);

        #pragma unroll
        for (int r = 0; r < 8; ++r)
            gates_lds[(r + rbase) * GP + gcol] = c[r];
        __syncthreads();

        // ---- layer-1 elementwise (torch gate order i,f,g,o): 2 cells / thread ----
        {
            float gi = gates_lds[em0 * GP +       eh0];
            float gf = gates_lds[em0 * GP +  64 + eh0];
            float gg = gates_lds[em0 * GP + 128 + eh0];
            float go = gates_lds[em0 * GP + 192 + eh0];
            float cn = sigf(gf) * c1r0 + sigf(gi) * tanhfast(gg);
            c1r0 = cn;
            h1_lds[em0 * HP + eh0] = (_Float16)(sigf(go) * tanhfast(cn));
            c1_lds[em0 * CP + eh0] = cn;

            gi = gates_lds[em1 * GP +       eh1];
            gf = gates_lds[em1 * GP +  64 + eh1];
            gg = gates_lds[em1 * GP + 128 + eh1];
            go = gates_lds[em1 * GP + 192 + eh1];
            cn = sigf(gf) * c1r1 + sigf(gi) * tanhfast(gg);
            c1r1 = cn;
            h1_lds[em1 * HP + eh1] = (_Float16)(sigf(go) * tanhfast(cn));
            c1_lds[em1 * CP + eh1] = cn;
        }
        __syncthreads();

        // ---- layer 2 gates (input = c1): all 512 threads, 8 MACs + shfl_xor butterfly ----
        {
            float4v ca = *(const float4v*)&c1_lds[dm * CP + dk];
            float4v cb = *(const float4v*)&c1_lds[dm * CP + dk + 4];
            float4v wa = *(const float4v*)&wih2_lds[dq * HID + dk];
            float4v wb = *(const float4v*)&wih2_lds[dq * HID + dk + 4];
            float acc = ca[0] * wa[0];
            acc = fmaf(ca[1], wa[1], acc);
            acc = fmaf(ca[2], wa[2], acc);
            acc = fmaf(ca[3], wa[3], acc);
            acc = fmaf(cb[0], wb[0], acc);
            acc = fmaf(cb[1], wb[1], acc);
            acc = fmaf(cb[2], wb[2], acc);
            acc = fmaf(cb[3], wb[3], acc);
            acc += __shfl_xor(acc, 1);
            acc += __shfl_xor(acc, 2);
            acc += __shfl_xor(acc, 4);
            if ((tid & 7) == 0)
                g2_lds[dm * 4 + dq] = acc + h2_lds[dm] * whh2_s + b2_s;
        }
        __syncthreads();

        // ---- layer-2 elementwise + output (emit c2); lanes 16..31 prefetch x(t+1) ----
        if (tid < ROWS) {
            float i2 = g2_lds[tid * 4 + 0];
            float f2 = g2_lds[tid * 4 + 1];
            float g2 = g2_lds[tid * 4 + 2];
            float o2 = g2_lds[tid * 4 + 3];
            float cn = sigf(f2) * c2r + sigf(i2) * tanhfast(g2);
            c2r = cn;
            h2_lds[tid] = sigf(o2) * tanhfast(cn);
            out[(brow + tid) * T + t] = cn;
        } else if (tid < 2 * ROWS && (t + 1) < T) {
            xt_lds[tid - ROWS] = x[(brow + tid - ROWS) * T + t + 1];
        }
        __syncthreads();
    }
}

extern "C" void kernel_launch(void* const* d_in, const int* in_sizes, int n_in,
                              void* d_out, int out_size, void* d_ws, size_t ws_size,
                              hipStream_t stream) {
    const float* x    = (const float*)d_in[0];
    const float* Wih1 = (const float*)d_in[1];
    const float* Whh1 = (const float*)d_in[2];
    const float* bih1 = (const float*)d_in[3];
    const float* bhh1 = (const float*)d_in[4];
    const float* Wih2 = (const float*)d_in[5];
    const float* Whh2 = (const float*)d_in[6];
    const float* bih2 = (const float*)d_in[7];
    const float* bhh2 = (const float*)d_in[8];
    float* out = (float*)d_out;

    const int T = in_sizes[0] / BATCH;   // x is [B,T], B=512 per reference
    dim3 grid(BATCH / ROWS);             // 32 workgroups, one per 16-row batch tile
    dim3 block(512);                     // 16 waves (wave32): one wave per 16 gate columns
    lstm2_persistent<<<grid, block, 0, stream>>>(x, Wih1, Whh1, bih1, bhh1,
                                                 Wih2, Whh2, bih2, bhh2, out, T);
}
